// GraphMemory_22067541967020
// MI455X (gfx1250) — compile-verified, matrix operand order checked by reading, not run on hardware
//
#include <hip/hip_runtime.h>
#include <hip/hip_bf16.h>

typedef __attribute__((ext_vector_type(16))) _Float16 v16h;
typedef __attribute__((ext_vector_type(8)))  _Float16 v8h;
typedef __attribute__((ext_vector_type(8)))  float    v8f;
typedef __attribute__((__vector_size__(16))) int      v4i;

#define GN 30000
#define GE 200000
#define GD 768
#define GR 10
#define GL 2
#define KREL (GR * GD)   // 7680

// CDNA5 async global->LDS fill (ASYNCcnt) if the toolchain exposes it.
#if defined(__has_builtin)
#if __has_builtin(__builtin_amdgcn_global_load_async_to_lds_b128) && \
    __has_builtin(__builtin_amdgcn_s_wait_asynccnt)
#define USE_ASYNC_LDS 1
#endif
#endif
#ifndef USE_ASYNC_LDS
#define USE_ASYNC_LDS 0
#endif

#if USE_ASYNC_LDS
#define AS_GLOBAL __attribute__((address_space(1)))
#define AS_LDS    __attribute__((address_space(3)))
// One 16B async copy: global (AS1) -> LDS (AS3), tracked by ASYNCcnt.
static __device__ __forceinline__ void async_cp16(const _Float16* g, _Float16* l) {
  __builtin_amdgcn_global_load_async_to_lds_b128(
      (AS_GLOBAL v4i*)(g), (AS_LDS v4i*)(l), 0, 0);
}
#endif

// ---------------- utility kernels ----------------

__global__ void k_zero(float* __restrict__ p, long n) {
  long i = (long)blockIdx.x * blockDim.x + threadIdx.x;
  if (i < n) p[i] = 0.0f;
}

// One block (256 threads) per edge: scatter-add h[src] into segment dst*R + etype.
__global__ __launch_bounds__(256) void k_scatter(
    const float* __restrict__ h, const int* __restrict__ ei /*[2,E]*/,
    const int* __restrict__ et, float* __restrict__ sums, float* __restrict__ cnts) {
  int e = blockIdx.x;
  int src = ei[e];
  int dst = ei[GE + e];
  int r   = et[e];
  long seg = (long)dst * GR + r;
  const float* hp = h + (long)src * GD;
  float* sp = sums + seg * GD;
  int t = threadIdx.x;
#pragma unroll
  for (int k = 0; k < 3; ++k)           // 768 / 256
    atomicAdd(&sp[t + k * 256], hp[t + k * 256]);
  if (t == 0) atomicAdd(&cnts[seg], 1.0f);
}

// means16[i] = (f16)(sums[i] / max(cnt,1))
__global__ void k_means16(const float* __restrict__ sums, const float* __restrict__ cnts,
                          _Float16* __restrict__ out, long n) {
  long i = (long)blockIdx.x * blockDim.x + threadIdx.x;
  if (i >= n) return;
  float c = cnts[i / GD];
  out[i] = (_Float16)(sums[i] / fmaxf(c, 1.0f));
}

__global__ void k_cvt16(const float* __restrict__ in, _Float16* __restrict__ out, long n) {
  long i = (long)blockIdx.x * blockDim.x + threadIdx.x;
  if (i < n) out[i] = (_Float16)in[i];
}

// W stored [K][GD] (k-major); emit transposed f16 [GD][K] so the B operand is
// N-major x K and shares the A-fragment LDS layout.
__global__ void k_transpose16(const float* __restrict__ Win, _Float16* __restrict__ out, int K) {
  long i = (long)blockIdx.x * blockDim.x + threadIdx.x;
  if (i >= (long)K * GD) return;
  int e = (int)(i / K);
  int k = (int)(i % K);
  out[i] = (_Float16)Win[(long)k * GD + e];
}

// ---------------- WMMA GEMM ----------------
// C[M,Ntot](f32) (+)= A[M,K](f16,row-major) * Bt[Ntot,K](f16,row-major)^T
// Block: 256 threads = 8 waves (wave32), tile 128(M) x 256(N), BK=32.
// Wave grid 2(M) x 4(N): each wave owns 64x64 -> 4x4 v_wmma_f32_16x16x32_f16.
// LDS double-buffered, one barrier per K-step. Stage fill is either
// global_load_async_to_lds_b128 (ASYNCcnt) or VGPR-staged loads + ds_store.

#define BM 128
#define BN 256
#define BK 32
#define LDT 40   // padded LDS row stride in f16 (80 B = 5*16 B: aligned + conflict-free)

union FragU { v16h v; v8h h[2]; };

__global__ __launch_bounds__(256) void k_gemm(
    const _Float16* __restrict__ A, const _Float16* __restrict__ Bt,
    float* __restrict__ C, int M, int K, int Ntot, int accumulate) {
  __shared__ _Float16 As[2][BM * LDT];   // 2 * 10240 B
  __shared__ _Float16 Bs[2][BN * LDT];   // 2 * 20480 B

  int tid = threadIdx.x;
  int wave = tid >> 5, lane = tid & 31;
  int hl = lane >> 4, l16 = lane & 15;      // K-half, row-in-16
  int waveM = wave >> 2, waveN = wave & 3;  // 2 x 4 wave grid
  int mBase = blockIdx.x * BM;
  int nBase = blockIdx.y * BN;

  // A tile 128x32: 2 threads/row, 16 f16 (two b128) each.
  int lrA = tid >> 1;
  int lcA = (tid & 1) << 4;
  int arow = mBase + lrA; if (arow >= M) arow = M - 1;   // clamp tail (stores guarded)
  const _Float16* agp = A + (size_t)arow * K + lcA;
  // B tile 256x32: 1 thread/row, 32 f16 (four b128) each.
  const _Float16* bgp = Bt + (size_t)(nBase + tid) * K;

#if USE_ASYNC_LDS
  // ---- prologue: stage 0 <- tile k0=0 (async DMA into LDS) ----
  {
    _Float16* aw = &As[0][lrA * LDT + lcA];
    _Float16* bw = &Bs[0][tid * LDT];
    async_cp16(agp,      aw);
    async_cp16(agp + 8,  aw + 8);
    async_cp16(bgp,      bw);
    async_cp16(bgp + 8,  bw + 8);
    async_cp16(bgp + 16, bw + 16);
    async_cp16(bgp + 24, bw + 24);
  }
  __builtin_amdgcn_s_wait_asynccnt(0);
  __syncthreads();
#else
  // ---- prologue: stage 0 <- tile k0=0 (VGPR-staged) ----
  v8h a0 = *(const v8h*)(agp);
  v8h a1 = *(const v8h*)(agp + 8);
  v8h b0 = *(const v8h*)(bgp);
  v8h b1 = *(const v8h*)(bgp + 8);
  v8h b2 = *(const v8h*)(bgp + 16);
  v8h b3 = *(const v8h*)(bgp + 24);
  {
    _Float16* aw = &As[0][lrA * LDT + lcA];
    _Float16* bw = &Bs[0][tid * LDT];
    *(v8h*)(aw)      = a0;
    *(v8h*)(aw + 8)  = a1;
    *(v8h*)(bw)      = b0;
    *(v8h*)(bw + 8)  = b1;
    *(v8h*)(bw + 16) = b2;
    *(v8h*)(bw + 24) = b3;
  }
  __syncthreads();
#endif

  v8f acc[4][4] = {};
  int s = 0;

  for (int k0 = 0; k0 < K; k0 += BK) {
    bool more = (k0 + BK) < K;
    if (more) {           // fill the other stage while this one computes
      agp += BK; bgp += BK;
#if USE_ASYNC_LDS
      _Float16* aw = &As[s ^ 1][lrA * LDT + lcA];
      _Float16* bw = &Bs[s ^ 1][tid * LDT];
      async_cp16(agp,      aw);
      async_cp16(agp + 8,  aw + 8);
      async_cp16(bgp,      bw);
      async_cp16(bgp + 8,  bw + 8);
      async_cp16(bgp + 16, bw + 16);
      async_cp16(bgp + 24, bw + 24);
#else
      a0 = *(const v8h*)(agp);
      a1 = *(const v8h*)(agp + 8);
      b0 = *(const v8h*)(bgp);
      b1 = *(const v8h*)(bgp + 8);
      b2 = *(const v8h*)(bgp + 16);
      b3 = *(const v8h*)(bgp + 24);
#endif
      if (k0 + 2 * BK < K) {          // CDNA5 global_prefetch_b8, tile after next
        __builtin_prefetch(agp + BK, 0, 3);
        __builtin_prefetch(bgp + BK, 0, 3);
      }
    }

    // Fragment layout (ISA 7.12.2, 16-bit 16x32): lane row = lane%16,
    // elements 0..7 -> k = hl*8+0..7, elements 8..15 -> k = 16+hl*8+0..7.
    // All 8 fragments hoisted into distinct registers so ds_load waits stagger
    // across the 16 WMMAs instead of stalling at dscnt==0 per j-group.
    const _Float16* asb = &As[s][0];
    const _Float16* bsb = &Bs[s][0];
    FragU af[4], bf[4];
#pragma unroll
    for (int i = 0; i < 4; ++i) {
      const _Float16* p = &asb[(waveM * 64 + i * 16 + l16) * LDT + hl * 8];
      af[i].h[0] = *(const v8h*)(p);
      af[i].h[1] = *(const v8h*)(p + 16);
    }
#pragma unroll
    for (int j = 0; j < 4; ++j) {
      const _Float16* p = &bsb[(waveN * 64 + j * 16 + l16) * LDT + hl * 8];
      bf[j].h[0] = *(const v8h*)(p);
      bf[j].h[1] = *(const v8h*)(p + 16);
    }
#pragma unroll
    for (int j = 0; j < 4; ++j)
#pragma unroll
      for (int i = 0; i < 4; ++i)
        acc[i][j] = __builtin_amdgcn_wmma_f32_16x16x32_f16(
            false, af[i].v, false, bf[j].v, (short)0, acc[i][j], false, false);

#if USE_ASYNC_LDS
    __builtin_amdgcn_s_wait_asynccnt(0);
#else
    if (more) {
      _Float16* aw = &As[s ^ 1][lrA * LDT + lcA];
      _Float16* bw = &Bs[s ^ 1][tid * LDT];
      *(v8h*)(aw)      = a0;
      *(v8h*)(aw + 8)  = a1;
      *(v8h*)(bw)      = b0;
      *(v8h*)(bw + 8)  = b1;
      *(v8h*)(bw + 16) = b2;
      *(v8h*)(bw + 24) = b3;
    }
#endif
    __syncthreads();
    s ^= 1;
  }

  // C/D layout: VGPR v -> m = v + 8*hl ; n = lane%16
#pragma unroll
  for (int i = 0; i < 4; ++i) {
    int rbase = mBase + waveM * 64 + i * 16 + hl * 8;
#pragma unroll
    for (int j = 0; j < 4; ++j) {
      int col = nBase + waveN * 64 + j * 16 + l16;
#pragma unroll
      for (int v = 0; v < 8; ++v) {
        int row = rbase + v;
        if (row < M) {
          size_t idx = (size_t)row * Ntot + col;
          float val = acc[i][j][v];
          if (accumulate) val += C[idx];
          C[idx] = val;
        }
      }
    }
  }
}

// ---------------- fused bias + LayerNorm + residual + ReLU ----------------
__global__ __launch_bounds__(256) void k_ln(
    const float* __restrict__ y, const float* __restrict__ bias,
    const float* __restrict__ g, const float* __restrict__ b,
    const float* __restrict__ resid, float* __restrict__ out, int do_relu) {
  __shared__ float sm[256];
  int n = blockIdx.x, t = threadIdx.x;
  const float* yr = y + (size_t)n * GD;
  float v[3];
  float s = 0.0f;
#pragma unroll
  for (int k = 0; k < 3; ++k) { v[k] = yr[t + k * 256] + bias[t + k * 256]; s += v[k]; }
  sm[t] = s; __syncthreads();
  for (int o = 128; o > 0; o >>= 1) { if (t < o) sm[t] += sm[t + o]; __syncthreads(); }
  float mu = sm[0] * (1.0f / GD);
  __syncthreads();
  float s2 = 0.0f;
#pragma unroll
  for (int k = 0; k < 3; ++k) { float d = v[k] - mu; s2 += d * d; }
  sm[t] = s2; __syncthreads();
  for (int o = 128; o > 0; o >>= 1) { if (t < o) sm[t] += sm[t + o]; __syncthreads(); }
  float rstd = rsqrtf(sm[0] * (1.0f / GD) + 1e-5f);
  float* orow = out + (size_t)n * GD;
  const float* rrow = resid ? resid + (size_t)n * GD : nullptr;
#pragma unroll
  for (int k = 0; k < 3; ++k) {
    int c = t + k * 256;
    float hv = (v[k] - mu) * rstd * g[c] + b[c];
    if (rrow) hv += rrow[c];
    if (do_relu) hv = fmaxf(hv, 0.0f);
    orow[c] = hv;
  }
}

// ---------------- host orchestration ----------------
extern "C" void kernel_launch(void* const* d_in, const int* in_sizes, int n_in,
                              void* d_out, int out_size, void* d_ws, size_t ws_size,
                              hipStream_t stream) {
  (void)in_sizes; (void)n_in; (void)out_size; (void)ws_size;
  const float* x      = (const float*)d_in[0];
  const int*   ei     = (const int*)d_in[1];   // [2,E]
  const int*   et     = (const int*)d_in[2];   // [E]
  const float* W_rel  = (const float*)d_in[3]; // [L,R,D,D]
  const float* W_root = (const float*)d_in[4]; // [L,D,D]
  const float* bias   = (const float*)d_in[5]; // [L,D]
  const float* ln_g   = (const float*)d_in[6];
  const float* ln_b   = (const float*)d_in[7];
  float* out = (float*)d_out;

  auto alignup = [](size_t v) { return (v + 255) & ~(size_t)255; };
  char* w = (char*)d_ws;
  float*    sums    = (float*)w;    w += alignup((size_t)GN * GR * GD * 4);
  float*    cnts    = (float*)w;    w += alignup((size_t)GN * GR * 4);
  _Float16* means16 = (_Float16*)w; w += alignup((size_t)GN * GR * GD * 2);
  _Float16* h16     = (_Float16*)w; w += alignup((size_t)GN * GD * 2);
  _Float16* WrelT   = (_Float16*)w; w += alignup((size_t)KREL * GD * 2);
  _Float16* WrootT  = (_Float16*)w; w += alignup((size_t)GD * GD * 2);
  float*    hbuf    = (float*)w;    w += alignup((size_t)GN * GD * 4);
  float*    ybuf    = (float*)w;    w += alignup((size_t)GN * GD * 4);

  const long nSums = (long)GN * GR * GD;
  const long nSeg  = (long)GN * GR;
  const long nH    = (long)GN * GD;
  dim3 gemmGrid((GN + BM - 1) / BM, GD / BN);

  for (int l = 0; l < GL; ++l) {
    const float* hin = (l == 0) ? x : hbuf;
    k_zero<<<(unsigned)((nSums + 255) / 256), 256, 0, stream>>>(sums, nSums);
    k_zero<<<(unsigned)((nSeg + 255) / 256), 256, 0, stream>>>(cnts, nSeg);
    k_scatter<<<GE, 256, 0, stream>>>(hin, ei, et, sums, cnts);
    k_means16<<<(unsigned)((nSums + 255) / 256), 256, 0, stream>>>(sums, cnts, means16, nSums);
    k_cvt16<<<(unsigned)((nH + 255) / 256), 256, 0, stream>>>(hin, h16, nH);
    k_transpose16<<<(unsigned)(((long)KREL * GD + 255) / 256), 256, 0, stream>>>(
        W_rel + (size_t)l * GR * GD * GD, WrelT, KREL);
    k_transpose16<<<(unsigned)(((long)GD * GD + 255) / 256), 256, 0, stream>>>(
        W_root + (size_t)l * GD * GD, WrootT, GD);
    // y = means @ W_rel  (M=30000, K=7680, N=768)
    k_gemm<<<gemmGrid, 256, 0, stream>>>(means16, WrelT, ybuf, GN, KREL, GD, 0);
    // y += h @ W_root    (K=768)
    k_gemm<<<gemmGrid, 256, 0, stream>>>(h16, WrootT, ybuf, GN, GD, GD, 1);
    float* hout = (l == GL - 1) ? out : hbuf;
    const float* resid = (l > 0) ? hin : nullptr;
    k_ln<<<GN, 256, 0, stream>>>(ybuf, bias + (size_t)l * GD, ln_g + (size_t)l * GD,
                                 ln_b + (size_t)l * GD, resid, hout, (l < GL - 1) ? 1 : 0);
  }
}
